// BondMatrixMessage_30726196036190
// MI455X (gfx1250) — matrix operand order repaired and checked by reading, not compiled, not run on hardware
//
#include <hip/hip_runtime.h>
#include <hip/hip_bf16.h>

// ---- problem constants (from reference) ----
#define BATCH    32
#define N_ATOMS  256
#define N_BONDS  512
#define DIM      64                    // ATOM_DIM == BOND_DIM == 64
#define E_TOTAL  (BATCH * N_BONDS)     // 16384 bonds
#define K_TOTAL  (DIM * DIM)           // 4096 reduction length (k*64+j)

typedef __attribute__((ext_vector_type(16))) __bf16 v16bf;
typedef __attribute__((ext_vector_type(8)))  float  v8f;
typedef __attribute__((ext_vector_type(4)))  int    v4i;

// ---- CDNA5 async global->LDS path (ASYNCcnt), with sync fallback ----
#if defined(__gfx1250__) && __has_builtin(__builtin_amdgcn_global_load_async_to_lds_b128)
#define HAVE_ASYNC_LDS 1
#else
#define HAVE_ASYNC_LDS 0
#endif

__device__ __forceinline__ void async_copy16(const void* g, void* l) {
#if HAVE_ASYNC_LDS
    __builtin_amdgcn_global_load_async_to_lds_b128((v4i*)g, (v4i*)l, 0, 0);
#else
    *(float4*)l = *(const float4*)g;
#endif
}

__device__ __forceinline__ void wait_async0() {
#if HAVE_ASYNC_LDS
#if __has_builtin(__builtin_amdgcn_s_wait_asynccnt)
    __builtin_amdgcn_s_wait_asynccnt(0);
#else
    asm volatile("s_wait_asynccnt 0x0" ::: "memory");
#endif
#endif
}

// ---------------------------------------------------------------------------
// Prep 1: Bt[i][k*64+j] = bond_transform[k][i*64+j], converted to bf16.
// ---------------------------------------------------------------------------
__global__ __launch_bounds__(256) void prep_bt_kernel(const float* __restrict__ W,
                                                      __bf16* __restrict__ Bt) {
    int idx = blockIdx.x * 256 + threadIdx.x;     // 64 * 4096 = 262144 total
    int i  = idx >> 12;
    int kj = idx & 4095;
    int k  = kj >> 6;
    int j  = kj & 63;
    Bt[idx] = (__bf16)W[k * K_TOTAL + i * DIM + j];
}

// ---------------------------------------------------------------------------
// Prep 2: gather source atoms per bond (connectivity is int64).
// ---------------------------------------------------------------------------
__global__ __launch_bounds__(256) void prep_src_kernel(const float* __restrict__ atom,
                                                       const long long* __restrict__ conn,
                                                       float* __restrict__ srcg) {
    int idx = blockIdx.x * 256 + threadIdx.x;     // E_TOTAL * 64 = 1048576
    int e = idx >> 6;
    int j = idx & 63;
    int b = e >> 9;
    long long a = conn[(long long)e * 2];
    srcg[idx] = atom[((long long)b * N_ATOMS + a) * DIM + j];
}

// ---------------------------------------------------------------------------
// Main GEMM: messages(16384x64) = X(16384x4096) * W2(4096x64)
//   X[e, k*64+j] = bond[e,k] * srcg[e,j]   (built in registers)
// 8 waves/block, 16 bonds/wave, full N=64 per wave (4 accum tiles).
// One LDS slab per k (64x64 bf16 = both j-halves) -> 1 barrier per 8 WMMAs.
// ---------------------------------------------------------------------------
__global__ __launch_bounds__(256) void bmm_gemm_kernel(const float*  __restrict__ bond,  // [E][64] fp32
                                                       const float*  __restrict__ srcg,  // [E][64] fp32
                                                       const __bf16* __restrict__ Bt,    // [64][4096] bf16
                                                       float* __restrict__ out) {        // [E][64] fp32
    __shared__ float sBond[128 * 64];                         // 32 KB bond tile
    __shared__ __align__(32) __bf16 sB[2][64 * 64];           // 2 x 8 KB slab, [n][kk0..63]

    const int tid  = threadIdx.x;
    const int wg   = blockIdx.x;
    const int wave = tid >> 5;
    const int lane = tid & 31;
    const int m    = lane & 15;     // A row / B,C column within 16-wide tile
    const int hb   = lane >> 4;     // lane half

    // --- bond tile for this block's 128 bonds: async global->LDS, 128B/thread
    {
        const char* g = (const char*)(bond + (size_t)wg * 128 * DIM);
        char* l = (char*)sBond;
        #pragma unroll
        for (int i = 0; i < 8; ++i) {
            int off = (tid + i * 256) * 16;
            async_copy16(g + off, l + off);
        }
    }

    // --- B slab loader: slab k = Bt[n][k*64 .. k*64+63] -> sB[buf][n*64+kk],
    //     32B per thread via two async b128 copies.
    const int bn  = tid >> 2;            // 0..63
    const int bk0 = (tid & 3) << 4;      // 0,16,32,48
    auto loadB = [&](int k, int buf) {
        const __bf16* g = Bt + (size_t)bn * K_TOTAL + k * DIM + bk0;
        __bf16* l = &sB[buf][bn * DIM + bk0];
        async_copy16(g, l);
        async_copy16(g + 8, l + 8);
        if (k + 2 < DIM)  // keep GL2 warm -> global_prefetch_b8
            __builtin_prefetch(g + 2 * DIM, 0, 1);
    };

    // --- per-lane src slices, pre-arranged in the A-fragment order:
    //     element p -> K = 16*hb + p (contiguous, matches B slab reads)
    const int e = wg * 128 + wave * 16 + m;
    const float* sp = srcg + (size_t)e * DIM;
    float s0[16], s1[16];
    #pragma unroll
    for (int t = 0; t < 16; ++t) {
        int jj = hb * 16 + t;
        s0[t] = sp[jj];            // j-half 0
        s1[t] = sp[32 + jj];       // j-half 1
    }

    v8f acc[4] = {v8f{0}, v8f{0}, v8f{0}, v8f{0}};

    loadB(0, 0);

    const int bondBase = (wave * 16 + m) * DIM;
    for (int k = 0; k < 64; ++k) {
        const int buf = k & 1;
        wait_async0();          // my async fills (issued last iter) complete
        __syncthreads();        // all waves filled slab; prior slab reads done
        if (k + 1 < 64) loadB(k + 1, buf ^ 1);

        const float bs = sBond[bondBase + k];
        const __bf16* Bbase = &sB[buf][0];

        #pragma unroll
        for (int jh = 0; jh < 2; ++jh) {
            const float* sv = jh ? s1 : s0;
            v16bf a;
            #pragma unroll
            for (int t = 0; t < 16; ++t) a[t] = (__bf16)(bs * sv[t]);

            const int kof = jh * 32 + hb * 16;
            v16bf b0 = *(const v16bf*)&Bbase[(0 * 16 + m) * DIM + kof];
            v16bf b1 = *(const v16bf*)&Bbase[(1 * 16 + m) * DIM + kof];
            v16bf b2 = *(const v16bf*)&Bbase[(2 * 16 + m) * DIM + kof];
            v16bf b3 = *(const v16bf*)&Bbase[(3 * 16 + m) * DIM + kof];

            acc[0] = __builtin_amdgcn_wmma_f32_16x16x32_bf16(false, a, false, b0, (short)0, acc[0], false, false);
            acc[1] = __builtin_amdgcn_wmma_f32_16x16x32_bf16(false, a, false, b1, (short)0, acc[1], false, false);
            acc[2] = __builtin_amdgcn_wmma_f32_16x16x32_bf16(false, a, false, b2, (short)0, acc[2], false, false);
            acc[3] = __builtin_amdgcn_wmma_f32_16x16x32_bf16(false, a, false, b3, (short)0, acc[3], false, false);
        }
    }

    // --- write out: C/D layout -> lane holds column m, rows hb*8 + r ---
    const int eRow = wg * 128 + wave * 16 + hb * 8;
    #pragma unroll
    for (int n = 0; n < 4; ++n)
        #pragma unroll
        for (int r = 0; r < 8; ++r)
            out[(size_t)(eRow + r) * DIM + n * 16 + m] = acc[n][r];
}

// ---------------------------------------------------------------------------
extern "C" void kernel_launch(void* const* d_in, const int* in_sizes, int n_in,
                              void* d_out, int out_size, void* d_ws, size_t ws_size,
                              hipStream_t stream) {
    const float*     atom = (const float*)d_in[0];       // (32,256,64) fp32
    const float*     bond = (const float*)d_in[1];       // (32,512,64) fp32
    const long long* conn = (const long long*)d_in[2];   // (32,512,2) int64
    const float*     W    = (const float*)d_in[3];       // (64,4096) fp32
    float*           out  = (float*)d_out;               // (32,512,64) fp32

    // workspace: srcg fp32 (4 MB) | Bt bf16 (512 KB)
    float*  srcg = (float*)d_ws;
    __bf16* Bt   = (__bf16*)((char*)d_ws + (size_t)E_TOTAL * DIM * sizeof(float));

    prep_bt_kernel <<<K_TOTAL * DIM / 256, 256, 0, stream>>>(W, Bt);
    prep_src_kernel<<<E_TOTAL * DIM / 256, 256, 0, stream>>>(atom, conn, srcg);
    bmm_gemm_kernel<<<E_TOTAL / 128, 256, 0, stream>>>(bond, srcg, Bt, out);
}
